// AuTopologyReadOut_37280316130041
// MI455X (gfx1250) — matrix operand order repaired and checked by reading, not compiled, not run on hardware
//
#include <hip/hip_runtime.h>
#include <hip/hip_bf16.h>
#include <math.h>

typedef __attribute__((ext_vector_type(16))) __bf16 v16bf;
typedef __attribute__((ext_vector_type(8)))  float  v8f;

__device__ __forceinline__ unsigned short bf16u(float x) {
  unsigned int u = __float_as_uint(x);
  u += 0x7FFFu + ((u >> 16) & 1u);   // round-to-nearest-even
  return (unsigned short)(u >> 16);
}

// Native CDNA5 tanh (V_TANH_F32) when available; otherwise a fast
// exp2/rcp-based evaluation: tanh(x) = 1 - 2/(e^{2x}+1).
__device__ __forceinline__ float fast_tanh(float x) {
#if __has_builtin(__builtin_amdgcn_tanhf)
  return __builtin_amdgcn_tanhf(x);
#else
  float t = __builtin_amdgcn_exp2f(x * 2.88539008177792681f);  // e^{2x}
  return 1.0f - 2.0f * __builtin_amdgcn_rcpf(t + 1.0f);
#endif
}

__device__ __forceinline__ v16bf ldfrag(const unsigned short* p) {
  return *(const v16bf*)p;
}

__device__ __forceinline__ v8f wmma(v16bf a, v16bf b, v8f c) {
  return __builtin_amdgcn_wmma_f32_16x16x32_bf16(false, a, false, b, (short)0, c,
                                                 false, false);
}

// ---------------------------------------------------------------------------
// Convert W[K x N] f32 (row-major) -> bf16 in WMMA B-fragment order:
// frag[((kt*NT + nt)*32 + lane)*16 + e] = W[kt*32 + (lane>>4)*8 + (e&7) + (e>>3)*16][nt*16 + (lane&15)]
// ---------------------------------------------------------------------------
__global__ void wconv_kernel(const float* __restrict__ W, unsigned short* __restrict__ frag,
                             int K, int N) {
  int idx = blockIdx.x * blockDim.x + threadIdx.x;
  if (idx >= K * N) return;
  int e    = idx & 15;
  int lane = (idx >> 4) & 31;
  int t    = idx >> 9;
  int NT   = N >> 4;
  int nt   = t % NT;
  int kt   = t / NT;
  int k = kt * 32 + (lane >> 4) * 8 + (e & 7) + (e >> 3) * 16;
  int n = nt * 16 + (lane & 15);
  frag[idx] = bf16u(W[k * N + n]);
}

__global__ void zero_kernel(float* __restrict__ out, int n) {
  int i = blockIdx.x * blockDim.x + threadIdx.x;
  if (i < n) out[i] = 0.0f;
}

// seg[start(m) .. start(m)+cnt[m]) = m
__global__ void segfill_kernel(const int* __restrict__ cnt, int* __restrict__ seg,
                               int nmax) {
  __shared__ int s_start;
  int m = blockIdx.x;
  if (threadIdx.x == 0) {
    int s = 0;
    for (int j = 0; j < m; ++j) s += cnt[j];
    s_start = s;
  }
  __syncthreads();
  int c = cnt[m];
  for (int i = threadIdx.x; i < c; i += blockDim.x) {
    int p = s_start + i;
    if (p < nmax) seg[p] = m;
  }
}

// ---------------------------------------------------------------------------
// Place a gathered feature into the 16-bit WMMA A-fragment layout in LDS.
// k within tile: kk = lh*8 + (e&7) + (e>>3)*16, lane = lh*16+row16.
// ---------------------------------------------------------------------------
__device__ __forceinline__ void stage_a(unsigned short* aF, int row16, int f, float v) {
  int kt = f >> 5;
  int kk = f & 31;
  int lh = (kk >> 3) & 1;
  int eh = kk >> 4;
  int e  = eh * 8 + (kk & 7);
  aF[(kt * 32 + (lh * 16 + row16)) * 16 + e] = bf16u(v);
}

// ===========================================================================
// Bond kernel: 32 bonds (two 16-row M-tiles) per workgroup, 8 waves.
// Dual MLP (r0, k).  Each B fragment feeds 2 WMMAs (one per M-tile); 4 WMMAs per k-step.
// ===========================================================================
__global__ void __launch_bounds__(256)
bond_kernel(const float* __restrict__ r, const float* __restrict__ xyz,
            const int* __restrict__ bonds, const int* __restrict__ seg,
            const unsigned short* __restrict__ w0A, const unsigned short* __restrict__ w1A,
            const float* __restrict__ b0A, const float* __restrict__ b1A,
            const float* __restrict__ w2A, const float* __restrict__ b2A,
            const unsigned short* __restrict__ w0B, const unsigned short* __restrict__ w1B,
            const float* __restrict__ b0B, const float* __restrict__ b1B,
            const float* __restrict__ w2B, const float* __restrict__ b2B,
            float* __restrict__ out, int nb) {
  __shared__ __attribute__((aligned(32))) unsigned short aF[2][8 * 32 * 16];      // 32x256 bf16
  __shared__ __attribute__((aligned(32))) unsigned short h1F[2][2][4 * 32 * 16];  // [mlp][tile] 16x128
  __shared__ float redu[2][32];
  __shared__ int bidx[32 * 2];

  const int tid  = threadIdx.x;
  const int wave = tid >> 5;
  const int lane = tid & 31;
  const int blk  = blockIdx.x;

  if (tid < 64) {
    int g = blk * 32 + (tid >> 1);
    bidx[tid] = (g < nb) ? bonds[g * 2 + (tid & 1)] : 0;
    redu[tid >> 5][tid & 31] = 0.0f;
  }
  __syncthreads();

  // ---- gather bond_in = r[b0] + r[b1] into A-fragment layout (coalesced) ----
  for (int i = 0; i < 32; ++i) {
    int a0 = bidx[i * 2], a1 = bidx[i * 2 + 1];
    float v = r[a0 * 256 + tid] + r[a1 * 256 + tid];
    stage_a(&aF[i >> 4][0], i & 15, tid, v);
  }
  __syncthreads();

  // ---- layer 1: [32x256] x [256x128], both MLPs ----
  v8f acc00 = {}, acc01 = {}, acc10 = {}, acc11 = {};
  for (int kt = 0; kt < 8; ++kt) {
    v16bf bA = ldfrag(&w0A[((kt * 8 + wave) * 32 + lane) * 16]);
    v16bf bB = ldfrag(&w0B[((kt * 8 + wave) * 32 + lane) * 16]);
    v16bf a0 = ldfrag(&aF[0][(kt * 32 + lane) * 16]);
    v16bf a1 = ldfrag(&aF[1][(kt * 32 + lane) * 16]);
    acc00 = wmma(a0, bA, acc00);  acc01 = wmma(a0, bB, acc01);
    acc10 = wmma(a1, bA, acc10);  acc11 = wmma(a1, bB, acc11);
  }
  {
    int n  = wave * 16 + (lane & 15);
    float bA = b0A[n], bB = b0B[n];
    int kt2 = wave >> 1;
    int eh  = wave & 1;
    int lh  = (lane >> 3) & 1;
    int e   = eh * 8 + (lane & 7);
    for (int v = 0; v < 8; ++v) {
      int m = v + 8 * (lane >> 4);
      int o = (kt2 * 32 + (lh * 16 + m)) * 16 + e;
      h1F[0][0][o] = bf16u(fast_tanh(acc00[v] + bA));
      h1F[1][0][o] = bf16u(fast_tanh(acc01[v] + bB));
      h1F[0][1][o] = bf16u(fast_tanh(acc10[v] + bA));
      h1F[1][1][o] = bf16u(fast_tanh(acc11[v] + bB));
    }
  }
  __syncthreads();

  // ---- layer 2: [32x128] x [128x128] ----
  v8f c00 = {}, c01 = {}, c10 = {}, c11 = {};
  for (int kt = 0; kt < 4; ++kt) {
    v16bf bA = ldfrag(&w1A[((kt * 8 + wave) * 32 + lane) * 16]);
    v16bf bB = ldfrag(&w1B[((kt * 8 + wave) * 32 + lane) * 16]);
    int o = (kt * 32 + lane) * 16;
    v16bf aA0 = ldfrag(&h1F[0][0][o]);
    v16bf aB0 = ldfrag(&h1F[1][0][o]);
    v16bf aA1 = ldfrag(&h1F[0][1][o]);
    v16bf aB1 = ldfrag(&h1F[1][1][o]);
    c00 = wmma(aA0, bA, c00);  c01 = wmma(aB0, bB, c01);
    c10 = wmma(aA1, bA, c10);  c11 = wmma(aB1, bB, c11);
  }
  // ---- layer 3: dot with W2[128x1], reduce across hidden ----
  {
    int n  = wave * 16 + (lane & 15);
    float bA = b1A[n], bB = b1B[n];
    float wA = w2A[n], wB = w2B[n];
    for (int v = 0; v < 8; ++v) {
      int m = v + 8 * (lane >> 4);
      atomicAdd(&redu[0][m],      fast_tanh(c00[v] + bA) * wA);
      atomicAdd(&redu[1][m],      fast_tanh(c01[v] + bB) * wB);
      atomicAdd(&redu[0][16 + m], fast_tanh(c10[v] + bA) * wA);
      atomicAdd(&redu[1][16 + m], fast_tanh(c11[v] + bB) * wB);
    }
  }
  __syncthreads();

  if (tid < 32) {
    int g = blk * 32 + tid;
    if (g < nb) {
      float mA = redu[0][tid] + b2A[0];
      float mB = redu[1][tid] + b2B[0];
      float r0 = 1.22474487139158905f + 0.1f * mA;  r0 = r0 * r0;  // (sqrt(1.5)+0.1m)^2
      float kb = 10.0f + mB;                        kb = kb * kb;  // (sqrt(100)+m)^2
      int i0 = bidx[tid * 2], i1 = bidx[tid * 2 + 1];
      float dx = xyz[i1 * 3 + 0] - xyz[i0 * 3 + 0];
      float dy = xyz[i1 * 3 + 1] - xyz[i0 * 3 + 1];
      float dz = xyz[i1 * 3 + 2] - xyz[i0 * 3 + 2];
      float d  = sqrtf(dx * dx + dy * dy + dz * dz);
      float df = d - r0;
      atomicAdd(&out[seg[g]], 0.5f * kb * df * df);
    }
  }
}

// ===========================================================================
// Angle kernel: 32 angles per workgroup, input dim 512, dual MLP (theta0, k)
// ===========================================================================
__global__ void __launch_bounds__(256)
angle_kernel(const float* __restrict__ r, const float* __restrict__ xyz,
             const int* __restrict__ angles, const int* __restrict__ seg,
             const unsigned short* __restrict__ w0A, const unsigned short* __restrict__ w1A,
             const float* __restrict__ b0A, const float* __restrict__ b1A,
             const float* __restrict__ w2A, const float* __restrict__ b2A,
             const unsigned short* __restrict__ w0B, const unsigned short* __restrict__ w1B,
             const float* __restrict__ b0B, const float* __restrict__ b1B,
             const float* __restrict__ w2B, const float* __restrict__ b2B,
             float* __restrict__ out, int na) {
  __shared__ __attribute__((aligned(32))) unsigned short aF[2][16 * 32 * 16];     // 32x512 bf16
  __shared__ __attribute__((aligned(32))) unsigned short h1F[2][2][4 * 32 * 16];  // [mlp][tile]
  __shared__ float redu[2][32];
  __shared__ int aidx[32 * 3];

  const int tid  = threadIdx.x;
  const int wave = tid >> 5;
  const int lane = tid & 31;
  const int blk  = blockIdx.x;

  if (tid < 96) {
    int g = blk * 32 + (tid / 3);
    aidx[tid] = (g < na) ? angles[g * 3 + (tid % 3)] : 0;
  }
  if (tid < 64) redu[tid >> 5][tid & 31] = 0.0f;
  __syncthreads();

  // ---- gather angle_in = concat(r[a0]+r[a2], r[a1]) into A fragments ----
  for (int i = 0; i < 64; ++i) {
    int idx = i * 256 + tid;
    int row = idx >> 9;
    int f   = idx & 511;
    int a0 = aidx[row * 3], a1 = aidx[row * 3 + 1], a2 = aidx[row * 3 + 2];
    float v = (f < 256) ? (r[a0 * 256 + f] + r[a2 * 256 + f])
                        : r[a1 * 256 + (f - 256)];
    stage_a(&aF[row >> 4][0], row & 15, f, v);
  }
  __syncthreads();

  // ---- layer 1: [32x512] x [512x128] ----
  v8f acc00 = {}, acc01 = {}, acc10 = {}, acc11 = {};
  for (int kt = 0; kt < 16; ++kt) {
    v16bf bA = ldfrag(&w0A[((kt * 8 + wave) * 32 + lane) * 16]);
    v16bf bB = ldfrag(&w0B[((kt * 8 + wave) * 32 + lane) * 16]);
    v16bf a0 = ldfrag(&aF[0][(kt * 32 + lane) * 16]);
    v16bf a1 = ldfrag(&aF[1][(kt * 32 + lane) * 16]);
    acc00 = wmma(a0, bA, acc00);  acc01 = wmma(a0, bB, acc01);
    acc10 = wmma(a1, bA, acc10);  acc11 = wmma(a1, bB, acc11);
  }
  {
    int n  = wave * 16 + (lane & 15);
    float bA = b0A[n], bB = b0B[n];
    int kt2 = wave >> 1;
    int eh  = wave & 1;
    int lh  = (lane >> 3) & 1;
    int e   = eh * 8 + (lane & 7);
    for (int v = 0; v < 8; ++v) {
      int m = v + 8 * (lane >> 4);
      int o = (kt2 * 32 + (lh * 16 + m)) * 16 + e;
      h1F[0][0][o] = bf16u(fast_tanh(acc00[v] + bA));
      h1F[1][0][o] = bf16u(fast_tanh(acc01[v] + bB));
      h1F[0][1][o] = bf16u(fast_tanh(acc10[v] + bA));
      h1F[1][1][o] = bf16u(fast_tanh(acc11[v] + bB));
    }
  }
  __syncthreads();

  // ---- layer 2 ----
  v8f c00 = {}, c01 = {}, c10 = {}, c11 = {};
  for (int kt = 0; kt < 4; ++kt) {
    v16bf bA = ldfrag(&w1A[((kt * 8 + wave) * 32 + lane) * 16]);
    v16bf bB = ldfrag(&w1B[((kt * 8 + wave) * 32 + lane) * 16]);
    int o = (kt * 32 + lane) * 16;
    v16bf aA0 = ldfrag(&h1F[0][0][o]);
    v16bf aB0 = ldfrag(&h1F[1][0][o]);
    v16bf aA1 = ldfrag(&h1F[0][1][o]);
    v16bf aB1 = ldfrag(&h1F[1][1][o]);
    c00 = wmma(aA0, bA, c00);  c01 = wmma(aB0, bB, c01);
    c10 = wmma(aA1, bA, c10);  c11 = wmma(aB1, bB, c11);
  }
  // ---- layer 3 ----
  {
    int n  = wave * 16 + (lane & 15);
    float bA = b1A[n], bB = b1B[n];
    float wA = w2A[n], wB = w2B[n];
    for (int v = 0; v < 8; ++v) {
      int m = v + 8 * (lane >> 4);
      atomicAdd(&redu[0][m],      fast_tanh(c00[v] + bA) * wA);
      atomicAdd(&redu[1][m],      fast_tanh(c01[v] + bB) * wB);
      atomicAdd(&redu[0][16 + m], fast_tanh(c10[v] + bA) * wA);
      atomicAdd(&redu[1][16 + m], fast_tanh(c11[v] + bB) * wB);
    }
  }
  __syncthreads();

  if (tid < 32) {
    int g = blk * 32 + tid;
    if (g < na) {
      float mA = redu[0][tid] + b2A[0];   // theta0 head
      float mB = redu[1][tid] + b2B[0];   // k head
      float t0 = 1.38243825f + mA;  t0 = t0 * t0;   // (sqrt(109.5*pi/180)+m)^2
      float ka = 3.16227766f + mB;  ka = ka * ka;   // (sqrt(10)+m)^2
      int i0 = aidx[tid * 3], i1 = aidx[tid * 3 + 1], i2 = aidx[tid * 3 + 2];
      float v1x = xyz[i1 * 3 + 0] - xyz[i0 * 3 + 0];
      float v1y = xyz[i1 * 3 + 1] - xyz[i0 * 3 + 1];
      float v1z = xyz[i1 * 3 + 2] - xyz[i0 * 3 + 2];
      float v2x = xyz[i2 * 3 + 0] - xyz[i1 * 3 + 0];
      float v2y = xyz[i2 * 3 + 1] - xyz[i1 * 3 + 1];
      float v2z = xyz[i2 * 3 + 2] - xyz[i1 * 3 + 2];
      float dot = -(v1x * v2x + v1y * v2y + v1z * v2z);
      float n1  = v1x * v1x + v1y * v1y + v1z * v1z;
      float n2  = v2x * v2x + v2y * v2y + v2z * v2z;
      float ct  = dot / sqrtf(n1 * n2);
      float th  = acosf(ct / 1.000001f);
      float df  = th - t0;
      atomicAdd(&out[seg[g]], 0.5f * ka * df * df);
    }
  }
}

// ===========================================================================
extern "C" void kernel_launch(void* const* d_in, const int* in_sizes, int n_in,
                              void* d_out, int out_size, void* d_ws, size_t ws_size,
                              hipStream_t stream) {
  const float* r      = (const float*)d_in[0];
  const float* xyz    = (const float*)d_in[1];
  const int*   bonds  = (const int*)d_in[2];
  const int*   angles = (const int*)d_in[3];
  const int*   numb   = (const int*)d_in[4];
  const int*   numa   = (const int*)d_in[5];
  // bond_r0_params
  const float* bW0r = (const float*)d_in[6];
  const float* bb0r = (const float*)d_in[7];
  const float* bW1r = (const float*)d_in[8];
  const float* bb1r = (const float*)d_in[9];
  const float* bW2r = (const float*)d_in[10];
  const float* bb2r = (const float*)d_in[11];
  // bond_k_params
  const float* bW0k = (const float*)d_in[12];
  const float* bb0k = (const float*)d_in[13];
  const float* bW1k = (const float*)d_in[14];
  const float* bb1k = (const float*)d_in[15];
  const float* bW2k = (const float*)d_in[16];
  const float* bb2k = (const float*)d_in[17];
  // angle_theta0_params
  const float* aW0t = (const float*)d_in[18];
  const float* ab0t = (const float*)d_in[19];
  const float* aW1t = (const float*)d_in[20];
  const float* ab1t = (const float*)d_in[21];
  const float* aW2t = (const float*)d_in[22];
  const float* ab2t = (const float*)d_in[23];
  // angle_k_params
  const float* aW0k = (const float*)d_in[24];
  const float* ab0k = (const float*)d_in[25];
  const float* aW1k = (const float*)d_in[26];
  const float* ab1k = (const float*)d_in[27];
  const float* aW2k = (const float*)d_in[28];
  const float* ab2k = (const float*)d_in[29];

  const int nb   = in_sizes[2] / 2;
  const int na   = in_sizes[3] / 3;
  const int nmol = in_sizes[4];
  float* out = (float*)d_out;

  // workspace layout: [bseg int nb][aseg int na][bf16 weight fragments]
  char* ws = (char*)d_ws;
  int* bseg = (int*)ws;
  int* aseg = bseg + nb;
  size_t woff = ((size_t)(nb + na) * sizeof(int) + 63) & ~(size_t)63;
  unsigned short* w = (unsigned short*)(ws + woff);
  unsigned short* f_bW0r = w;  w += 256 * 128;
  unsigned short* f_bW0k = w;  w += 256 * 128;
  unsigned short* f_bW1r = w;  w += 128 * 128;
  unsigned short* f_bW1k = w;  w += 128 * 128;
  unsigned short* f_aW0t = w;  w += 512 * 128;
  unsigned short* f_aW0k = w;  w += 512 * 128;
  unsigned short* f_aW1t = w;  w += 128 * 128;
  unsigned short* f_aW1k = w;  w += 128 * 128;

  zero_kernel<<<(nmol + 255) / 256, 256, 0, stream>>>(out, nmol);
  segfill_kernel<<<nmol, 128, 0, stream>>>(numb, bseg, nb);
  segfill_kernel<<<nmol, 128, 0, stream>>>(numa, aseg, na);

  wconv_kernel<<<(256 * 128 + 255) / 256, 256, 0, stream>>>(bW0r, f_bW0r, 256, 128);
  wconv_kernel<<<(256 * 128 + 255) / 256, 256, 0, stream>>>(bW0k, f_bW0k, 256, 128);
  wconv_kernel<<<(128 * 128 + 255) / 256, 256, 0, stream>>>(bW1r, f_bW1r, 128, 128);
  wconv_kernel<<<(128 * 128 + 255) / 256, 256, 0, stream>>>(bW1k, f_bW1k, 128, 128);
  wconv_kernel<<<(512 * 128 + 255) / 256, 256, 0, stream>>>(aW0t, f_aW0t, 512, 128);
  wconv_kernel<<<(512 * 128 + 255) / 256, 256, 0, stream>>>(aW0k, f_aW0k, 512, 128);
  wconv_kernel<<<(128 * 128 + 255) / 256, 256, 0, stream>>>(aW1t, f_aW1t, 128, 128);
  wconv_kernel<<<(128 * 128 + 255) / 256, 256, 0, stream>>>(aW1k, f_aW1k, 128, 128);

  bond_kernel<<<(nb + 31) / 32, 256, 0, stream>>>(
      r, xyz, bonds, bseg,
      f_bW0r, f_bW1r, bb0r, bb1r, bW2r, bb2r,
      f_bW0k, f_bW1k, bb0k, bb1k, bW2k, bb2k,
      out, nb);

  angle_kernel<<<(na + 31) / 32, 256, 0, stream>>>(
      r, xyz, angles, aseg,
      f_aW0t, f_aW1t, ab0t, ab1t, aW2t, ab2t,
      f_aW0k, f_aW1k, ab0k, ab1k, aW2k, ab2k,
      out, na);
}
